// OptimizedVQA_91233695301889
// MI455X (gfx1250) — matrix operand order repaired and checked by reading, not compiled, not run on hardware
//
#include <hip/hip_runtime.h>

// ---------------------------------------------------------------------------
// Types for CDNA5 WMMA (gfx1250, wave32)
// ---------------------------------------------------------------------------
typedef float  v8f   __attribute__((ext_vector_type(8)));
typedef __bf16 v16bf __attribute__((ext_vector_type(16)));

__device__ __forceinline__ unsigned short f2bf(float x) {
  union { float f; unsigned u; } v; v.f = x;
  unsigned r = v.u + 0x7FFFu + ((v.u >> 16) & 1u);   // round-to-nearest-even
  return (unsigned short)(r >> 16);
}
__device__ __forceinline__ float bf2f(unsigned short h) {
  union { unsigned u; float f; } v; v.u = ((unsigned)h) << 16; return v.f;
}
__device__ __forceinline__ float gelu_f(float x) {   // exact GELU (approximate=False)
  return 0.5f * x * (1.0f + erff(x * 0.70710678118654752440f));
}

// ---------------------------------------------------------------------------
// Fused WMMA GEMM:  Out = Epilogue( A[M,K] @ Wt[N,K]^T + bias )
//   - A: row-major, element stride lda, column offset a_col (fp32 or bf16)
//   - Wt: bf16, row-major [N,K] (i.e. W transposed), L2-resident
//   - DO_LN:      LayerNorm(over full N=768 row) -> GELU -> bf16 store
//   - FUSION_ADD: add row_add[qt[row]*768 + col] before LN (fusion layer)
//   - !DO_LN:     bias -> GELU -> bf16 store (head hidden layers)
// Block: 256 threads = 8 waves = 2 (M halves of 16) x 4 (N columns of 16*NT).
// Block tile: 32 rows x (64*NT) cols. K loop step 32, double-buffered LDS A.
// ---------------------------------------------------------------------------
template <int NT, bool DO_LN, bool FUSION_ADD, bool A_F32>
__global__ __launch_bounds__(256) void vqa_wmma_gemm(
    const void* __restrict__ Av, int lda, int a_col,
    const unsigned short* __restrict__ Wt,
    const float* __restrict__ bias,
    const float* __restrict__ gamma,
    const float* __restrict__ beta,
    const float* __restrict__ row_add,
    const int* __restrict__ qt,
    unsigned short* __restrict__ Out, int ldo,
    int K)
{
  constexpr int BN = NT * 64;   // block N coverage
  constexpr int WN = NT * 16;   // wave  N coverage

  // A tile 32x32 bf16, padded stride 40 elems (80B: 16B-aligned rows,
  // r*20 DWORDs mod 64 distinct for 16 lanes -> conflict-free b128 reads)
  __shared__ unsigned short As[2][32 * 40];
  __shared__ float red_s[32][4];
  __shared__ float red_q[32][4];

  const int tid   = threadIdx.x;
  const int wave  = tid >> 5;
  const int lane  = tid & 31;
  const int m0    = blockIdx.x * 32;
  const int nblk  = blockIdx.y * BN;
  const int wm    = (wave & 4) ? 16 : 0;
  const int wn    = (wave & 3) * WN;
  const int lhalf = (lane < 16) ? 0 : 1;
  const int l15   = lane & 15;

  v8f acc[NT];
#pragma unroll
  for (int t = 0; t < NT; ++t)
#pragma unroll
    for (int j = 0; j < 8; ++j) acc[t][j] = 0.0f;

  // --- A-tile stage loader: 32x32 elems, 4 per thread ---
  const int sr = tid >> 3;        // row 0..31
  const int sc = (tid & 7) * 4;   // col 0,4,...,28
  auto stage = [&](int buf, int kk) {
    if (A_F32) {
      const float* A = (const float*)Av;
      float4 f = *(const float4*)(A + (size_t)(m0 + sr) * lda + a_col + kk + sc);
      uint2 p;
      p.x = (unsigned)f2bf(f.x) | ((unsigned)f2bf(f.y) << 16);
      p.y = (unsigned)f2bf(f.z) | ((unsigned)f2bf(f.w) << 16);
      *(uint2*)&As[buf][sr * 40 + sc] = p;
    } else {
      const unsigned short* A = (const unsigned short*)Av;
      uint2 d = *(const uint2*)(A + (size_t)(m0 + sr) * lda + a_col + kk + sc);
      *(uint2*)&As[buf][sr * 40 + sc] = d;
    }
  };

  stage(0, 0);
  __syncthreads();

  // ISA 7.12.2 fragment maps (16-bit A 16x32; B 32x16):
  //   A lane<16: K runs [0,8)+[16,24);  lane>=16: [8,16)+[24,32)
//   B lane<16: K [0,8)+[16,24);       lane>=16: [8,16)+[24,32)
  const int arow = wm + l15;
  const int asel = lhalf ? 8 : 0;
  const int bsel = lhalf ? 8 : 0;

  for (int kk = 0; kk < K; kk += 32) {
    const int cur = (kk >> 5) & 1;
    if (kk + 32 < K) stage(cur ^ 1, kk + 32);

    union { uint4 q[2]; v16bf v; } af;
    const unsigned short* ab = &As[cur][arow * 40 + asel];
    af.q[0] = *(const uint4*)(ab);
    af.q[1] = *(const uint4*)(ab + 16);

#pragma unroll
    for (int t = 0; t < NT; ++t) {
      const int n = nblk + wn + t * 16 + l15;
      const unsigned short* bb = Wt + (size_t)n * K + kk + bsel;
      union { uint4 q[2]; v16bf v; } bfr;
      bfr.q[0] = *(const uint4*)(bb);
      bfr.q[1] = *(const uint4*)(bb + 16);
      acc[t] = __builtin_amdgcn_wmma_f32_16x16x32_bf16(
          false, af.v, false, bfr.v, (short)0, acc[t], false, false);
    }
    __syncthreads();
  }

  // --- Epilogue ---
  // C/D layout: vgpr j, lane l -> row = (l<16 ? j : j+8), col = l&15
  if (DO_LN) {
    float s[8], s2[8];
#pragma unroll
    for (int j = 0; j < 8; ++j) { s[j] = 0.f; s2[j] = 0.f; }
    int qj[8];
    if (FUSION_ADD) {
#pragma unroll
      for (int j = 0; j < 8; ++j) qj[j] = qt[m0 + wm + lhalf * 8 + j];
    }
#pragma unroll
    for (int t = 0; t < NT; ++t) {
      const int c = nblk + wn + t * 16 + l15;
      const float bv = bias[c];
#pragma unroll
      for (int j = 0; j < 8; ++j) {
        float v = acc[t][j] + bv;
        if (FUSION_ADD) v += row_add[(size_t)qj[j] * 768 + c];
        acc[t][j] = v;
        s[j]  += v;
        s2[j] += v * v;
      }
    }
    // reduce within each 16-lane half (xor masks < 16 stay in-half on wave32)
#pragma unroll
    for (int m = 1; m < 16; m <<= 1) {
#pragma unroll
      for (int j = 0; j < 8; ++j) {
        s[j]  += __shfl_xor(s[j],  m);
        s2[j] += __shfl_xor(s2[j], m);
      }
    }
    if (l15 == 0) {
      const int wc = wave & 3;
#pragma unroll
      for (int j = 0; j < 8; ++j) {
        const int r = wm + lhalf * 8 + j;
        red_s[r][wc] = s[j];
        red_q[r][wc] = s2[j];
      }
    }
    __syncthreads();
    float mean[8], rstd[8];
#pragma unroll
    for (int j = 0; j < 8; ++j) {
      const int r = wm + lhalf * 8 + j;
      const float S = red_s[r][0] + red_s[r][1] + red_s[r][2] + red_s[r][3];
      const float Q = red_q[r][0] + red_q[r][1] + red_q[r][2] + red_q[r][3];
      const float mu = S * (1.0f / 768.0f);
      mean[j] = mu;
      rstd[j] = rsqrtf(Q * (1.0f / 768.0f) - mu * mu + 1e-5f);
    }
#pragma unroll
    for (int t = 0; t < NT; ++t) {
      const int c = nblk + wn + t * 16 + l15;
      const float g = gamma[c], be = beta[c];
#pragma unroll
      for (int j = 0; j < 8; ++j) {
        const int r = m0 + wm + lhalf * 8 + j;
        const float y = (acc[t][j] - mean[j]) * rstd[j] * g + be;
        Out[(size_t)r * ldo + c] = f2bf(gelu_f(y));
      }
    }
  } else {
#pragma unroll
    for (int t = 0; t < NT; ++t) {
      const int c = nblk + wn + t * 16 + l15;
      const float bv = bias[c];
#pragma unroll
      for (int j = 0; j < 8; ++j) {
        const int r = m0 + wm + lhalf * 8 + j;
        Out[(size_t)r * ldo + c] = f2bf(gelu_f(acc[t][j] + bv));
      }
    }
  }
}

// ---------------------------------------------------------------------------
// Weight prep: fp32 [K,N] -> bf16 transposed [N,K]
// ---------------------------------------------------------------------------
__global__ void vqa_transpose_cvt(const float* __restrict__ src,
                                  unsigned short* __restrict__ dst,
                                  int K, int N) {
  const int idx = blockIdx.x * blockDim.x + threadIdx.x;
  if (idx >= K * N) return;
  const int n = idx / K, k = idx % K;
  dst[(size_t)n * K + k] = f2bf(src[(size_t)k * N + n]);
}

// type_contrib[t][n] = sum_j temb[t][j] * Wf[(768+j)*768 + n]   (4 x 768)
__global__ void vqa_type_contrib(const float* __restrict__ temb,
                                 const float* __restrict__ Wf,
                                 float* __restrict__ tc) {
  const int idx = blockIdx.x * blockDim.x + threadIdx.x;
  if (idx >= 4 * 768) return;
  const int t = idx / 768, n = idx % 768;
  float a = 0.f;
  for (int j = 0; j < 192; ++j)
    a += temb[t * 192 + j] * Wf[(size_t)(768 + j) * 768 + n];
  tc[idx] = a;
}

// combined head-layer-1 bias [1280] = [color(256) shape(256) count(256) spatial(512)]
__global__ void vqa_head_bias(const float* cb, const float* sb,
                              const float* nb, const float* pb,
                              float* __restrict__ bh) {
  const int i = blockIdx.x * blockDim.x + threadIdx.x;
  if (i >= 1280) return;
  float v;
  if (i < 256) v = cb[i];
  else if (i < 512) v = sb[i - 256];
  else if (i < 768) v = nb[i - 512];
  else v = pb[i - 768];
  bh[i] = v;
}

// ---------------------------------------------------------------------------
// Final routing: per sample, compute only the selected head's logits, pad -1e9
// thread = (row, j) with 16 j-slots (j<13 active)
// ---------------------------------------------------------------------------
__global__ void vqa_route(const unsigned short* __restrict__ HH,   // [B,1280] bf16
                          const unsigned short* __restrict__ S2,   // [B,256]  bf16
                          const int* __restrict__ qt,
                          const float* __restrict__ cw2, const float* __restrict__ cb2,
                          const float* __restrict__ sw2, const float* __restrict__ sb2,
                          const float* __restrict__ nw2, const float* __restrict__ nb2,
                          const float* __restrict__ pw3, const float* __restrict__ pb3,
                          float* __restrict__ out, int B) {
  const int idx = blockIdx.x * blockDim.x + threadIdx.x;
  const int j = idx & 15;
  const int row = idx >> 4;
  if (row >= B || j >= 13) return;
  const int q = qt[row];
  float r;
  if (q == 3) {
    const unsigned short* x = S2 + (size_t)row * 256;
    float a = pb3[j];
    for (int k = 0; k < 256; ++k) a += bf2f(x[k]) * pw3[k * 13 + j];
    r = a;
  } else {
    const int nc = (q == 1) ? 3 : 4;
    if (j >= nc) {
      r = -1e9f;
    } else {
      const float *W, *bb;
      if (q == 0)      { W = cw2; bb = cb2; }
      else if (q == 1) { W = sw2; bb = sb2; }
      else             { W = nw2; bb = nb2; }
      const unsigned short* x = HH + (size_t)row * 1280 + q * 256;
      float a = bb[j];
      for (int k = 0; k < 256; ++k) a += bf2f(x[k]) * W[k * nc + j];
      r = a;
    }
  }
  out[(size_t)row * 13 + j] = r;
}

// ---------------------------------------------------------------------------
// Host launcher
// ---------------------------------------------------------------------------
extern "C" void kernel_launch(void* const* d_in, const int* in_sizes, int n_in,
                              void* d_out, int out_size, void* d_ws, size_t ws_size,
                              hipStream_t stream) {
  (void)n_in; (void)out_size; (void)ws_size;
  const int B = 65536;

  // ---- input index maps (handle sorted-key vs insertion-order flattening) ----
  int i_cw1, i_cb1, i_cw2, i_cb2;            // color
  int i_nw1, i_nb1, i_nw2, i_nb2;            // count
  int i_fw, i_fb, i_fg, i_fbe;               // fusion
  int i_sw1, i_sb1, i_sw2, i_sb2;            // shape
  int i_pw1, i_pb1, i_pw2, i_pb2, i_pw3, i_pb3; // spatial
  int i_te;                                   // type_emb
  int i_v1w, i_v1b, i_v1g, i_v1be;
  int i_v2w, i_v2b, i_v2g, i_v2be;
  int i_v3w, i_v3b, i_v3g, i_v3be;

  if (in_sizes[2] == 512 * 768) {
    // insertion order: vm, type_emb, fusion, color, shape, count, spatial
    i_v1w=2;  i_v1b=3;  i_v1g=4;  i_v1be=5;
    i_v2w=6;  i_v2b=7;  i_v2g=8;  i_v2be=9;
    i_v3w=10; i_v3b=11; i_v3g=12; i_v3be=13;
    i_te=14;
    i_fw=15; i_fb=16; i_fg=17; i_fbe=18;
    i_cw1=19; i_cb1=20; i_cw2=21; i_cb2=22;
    i_sw1=23; i_sb1=24; i_sw2=25; i_sb2=26;
    i_nw1=27; i_nb1=28; i_nw2=29; i_nb2=30;
    i_pw1=31; i_pb1=32; i_pw2=33; i_pb2=34; i_pw3=35; i_pb3=36;
  } else {
    // sorted keys: color, count, fusion, shape, spatial, type_emb, vm
    i_cw1=2;  i_cb1=3;  i_cw2=4;  i_cb2=5;
    i_nw1=6;  i_nb1=7;  i_nw2=8;  i_nb2=9;
    i_fw=10; i_fb=11; i_fg=12; i_fbe=13;
    i_sw1=14; i_sb1=15; i_sw2=16; i_sb2=17;
    i_pw1=18; i_pb1=19; i_pw2=20; i_pb2=21; i_pw3=22; i_pb3=23;
    i_te=24;
    i_v1w=25; i_v1b=26; i_v1g=27; i_v1be=28;
    i_v2w=29; i_v2b=30; i_v2g=31; i_v2be=32;
    i_v3w=33; i_v3b=34; i_v3g=35; i_v3be=36;
  }

  const float* vis = (const float*)d_in[0];
  const int*   qt  = (const int*)d_in[1];
  #define F(i) ((const float*)d_in[i])

  // ---- workspace carve ----
  char* p = (char*)d_ws;
  auto carve = [&](size_t bytes) { char* r = p; p += (bytes + 255) & ~(size_t)255; return r; };
  unsigned short* Ha   = (unsigned short*)carve((size_t)B * 768 * 2);
  unsigned short* Hb   = (unsigned short*)carve((size_t)B * 768 * 2);
  unsigned short* HH   = (unsigned short*)carve((size_t)B * 1280 * 2);
  unsigned short* S2   = (unsigned short*)carve((size_t)B * 256 * 2);
  unsigned short* W1t  = (unsigned short*)carve((size_t)768 * 512 * 2);
  unsigned short* W2t  = (unsigned short*)carve((size_t)768 * 768 * 2);
  unsigned short* W3t  = (unsigned short*)carve((size_t)768 * 768 * 2);
  unsigned short* Wft  = (unsigned short*)carve((size_t)768 * 768 * 2);
  unsigned short* WHt  = (unsigned short*)carve((size_t)1280 * 768 * 2);
  unsigned short* Wp2t = (unsigned short*)carve((size_t)256 * 512 * 2);
  float*          bh   = (float*)carve(1280 * 4);
  float*          tc   = (float*)carve(4 * 768 * 4);

  // ---- weight prep (tiny, L2-resident) ----
  auto tcvt = [&](const float* src, unsigned short* dst, int K, int N) {
    const int total = K * N;
    vqa_transpose_cvt<<<(total + 255) / 256, 256, 0, stream>>>(src, dst, K, N);
  };
  tcvt(F(i_v1w), W1t, 512, 768);
  tcvt(F(i_v2w), W2t, 768, 768);
  tcvt(F(i_v3w), W3t, 768, 768);
  tcvt(F(i_fw),  Wft, 768, 768);              // first 768 rows of [960,768]
  tcvt(F(i_cw1), WHt + (size_t)0 * 768,   768, 256);
  tcvt(F(i_sw1), WHt + (size_t)256 * 768, 768, 256);
  tcvt(F(i_nw1), WHt + (size_t)512 * 768, 768, 256);
  tcvt(F(i_pw1), WHt + (size_t)768 * 768, 768, 512);
  tcvt(F(i_pw2), Wp2t, 512, 256);
  vqa_type_contrib<<<12, 256, 0, stream>>>(F(i_te), F(i_fw), tc);
  vqa_head_bias<<<5, 256, 0, stream>>>(F(i_cb1), F(i_sb1), F(i_nb1), F(i_pb1), bh);

  // ---- pipeline ----
  dim3 g1(B / 32, 1);
  // L1: vis(f32)[B,512] -> Ha
  vqa_wmma_gemm<12, true, false, true><<<g1, 256, 0, stream>>>(
      vis, 512, 0, W1t, F(i_v1b), F(i_v1g), F(i_v1be), nullptr, nullptr, Ha, 768, 512);
  // L2: Ha -> Hb
  vqa_wmma_gemm<12, true, false, false><<<g1, 256, 0, stream>>>(
      Ha, 768, 0, W2t, F(i_v2b), F(i_v2g), F(i_v2be), nullptr, nullptr, Hb, 768, 768);
  // L3: Hb -> Ha
  vqa_wmma_gemm<12, true, false, false><<<g1, 256, 0, stream>>>(
      Hb, 768, 0, W3t, F(i_v3b), F(i_v3g), F(i_v3be), nullptr, nullptr, Ha, 768, 768);
  // fusion: Ha (+type_contrib[q]) -> Hb
  vqa_wmma_gemm<12, true, true, false><<<g1, 256, 0, stream>>>(
      Ha, 768, 0, Wft, F(i_fb), F(i_fg), F(i_fbe), tc, qt, Hb, 768, 768);
  // heads layer 1 (combined 768 -> 1280, GELU): Hb -> HH
  dim3 g2(B / 32, 5);
  vqa_wmma_gemm<4, false, false, false><<<g2, 256, 0, stream>>>(
      Hb, 768, 0, WHt, bh, nullptr, nullptr, nullptr, nullptr, HH, 1280, 768);
  // spatial layer 2 (512 -> 256, GELU): HH[:,768:1280] -> S2
  dim3 g3(B / 32, 1);
  vqa_wmma_gemm<4, false, false, false><<<g3, 256, 0, stream>>>(
      HH, 1280, 768, Wp2t, F(i_pb2), nullptr, nullptr, nullptr, nullptr, S2, 256, 512);
  // routing + final tiny layers
  vqa_route<<<(B * 16) / 256, 256, 0, stream>>>(
      HH, S2, qt, F(i_cw2), F(i_cb2), F(i_sw2), F(i_sb2), F(i_nw2), F(i_nb2),
      F(i_pw3), F(i_pb3), (float*)d_out, B);
  #undef F
}
